// GATDummy_2345052144336
// MI455X (gfx1250) — compile-verified, matrix operand order checked by reading, not run on hardware
//
#include <hip/hip_runtime.h>
#include <hip/hip_bf16.h>
#include <math.h>

typedef __attribute__((ext_vector_type(16))) __bf16 v16bf;
typedef __attribute__((ext_vector_type(8)))  float  v8f;

#define NEG_SLOPE 0.2f

// ---------- helpers ----------
__device__ __forceinline__ void atomicMaxF(float* addr, float v) {
  // monotone ordered-int trick; correct when initialized to -inf
  if (v >= 0.0f) atomicMax((int*)addr, __float_as_int(v));
  else           atomicMin((unsigned int*)addr, __float_as_uint(v));
}

// ---------- f32 -> bf16 (optional fused ELU) ----------
__global__ void k_cvt_bf16(const float* __restrict__ in, __bf16* __restrict__ out,
                           long n, int elu) {
  long i = (long)blockIdx.x * blockDim.x + threadIdx.x;
  if (i >= n) return;
  float v = in[i];
  if (elu) v = v > 0.0f ? v : expm1f(v);
  out[i] = (__bf16)v;
}

// ---------- pack W [K,Nc] row-major f32 -> [NcPad][K] col-major bf16 ----------
__global__ void k_pack_w(const float* __restrict__ W, __bf16* __restrict__ Bp,
                         int K, int Nc, int NcPad) {
  long total = (long)NcPad * K;
  long i = (long)blockIdx.x * blockDim.x + threadIdx.x;
  if (i >= total) return;
  int n = (int)(i / K);
  int k = (int)(i - (long)n * K);
  float v = (n < Nc) ? W[(long)k * Nc + n] : 0.0f;
  Bp[i] = (__bf16)v;
}

// ---------- WMMA GEMM: C[M,Nc] = A[M,K](bf16 rowmajor) x Bp(colmajor packed) ----------
// One wave per block (blockDim == 32). Each wave computes a 16 x (NT*16) strip.
// Grid is exact (Mtiles * nGroups blocks) so EXEC is all-ones at every WMMA.
template <int NT>
__global__ void k_gemm_wmma(const __bf16* __restrict__ A, const __bf16* __restrict__ Bp,
                            float* __restrict__ C, int K, int Nc, int nGroups) {
  int wave = blockIdx.x;                    // scalar (SGPR)
  int lane = threadIdx.x;                   // 0..31
  int mi = wave / nGroups;
  int ng = wave - mi * nGroups;

  int l16   = lane & 15;
  int khalf = (lane >> 4) << 4;             // lanes 16..31 hold K+16..K+31

  const __bf16* arow  = A  + (long)(mi * 16 + l16) * K + khalf;
  const __bf16* bbase = Bp + (long)(ng * NT * 16 + l16) * K + khalf;

  v8f acc[NT] = {};
  for (int k0 = 0; k0 < K; k0 += 32) {
    v16bf a = *(const v16bf*)(arow + k0);
#pragma unroll
    for (int t = 0; t < NT; ++t) {
      v16bf b = *(const v16bf*)(bbase + (long)t * 16 * K + k0);
      acc[t] = __builtin_amdgcn_wmma_f32_16x16x32_bf16(
          false, a, false, b, (short)0, acc[t], false, false);
    }
  }

  // store: vgpr i -> row mi*16 + i + (lane>=16 ? 8 : 0), col = tile*16 + (lane&15)
  int mrow = mi * 16 + ((lane >> 4) << 3);
#pragma unroll
  for (int t = 0; t < NT; ++t) {
    int n = (ng * NT + t) * 16 + l16;
    if (n < Nc) {                           // lane-level store mask (layer 5 padding)
#pragma unroll
      for (int i = 0; i < 8; ++i)
        C[(long)(mrow + i) * Nc + n] = acc[t][i];
    }
  }
}

// ---------- per-node attention scores: el/er[n,h] = sum_f z[n,h,f]*a{l,r}[h,f] ----------
__global__ void k_node_scores(const float* __restrict__ z, const float* __restrict__ al,
                              const float* __restrict__ ar, float* __restrict__ el,
                              float* __restrict__ er, long N, int H, int F) {
  long i = (long)blockIdx.x * blockDim.x + threadIdx.x;
  if (i >= N * H) return;
  long n = i / H;
  int h = (int)(i - n * H);
  const float* zp  = z + (n * H + h) * (long)F;
  const float* alp = al + (long)h * F;
  const float* arp = ar + (long)h * F;
  float sl = 0.0f, sr = 0.0f;
  for (int f = 0; f < F; ++f) { float v = zp[f]; sl += v * alp[f]; sr += v * arp[f]; }
  el[i] = sl; er[i] = sr;
}

// ---------- init m=-inf, s=0 ----------
__global__ void k_init_ms(float* __restrict__ m, float* __restrict__ s, long n) {
  long i = (long)blockIdx.x * blockDim.x + threadIdx.x;
  if (i < n) { m[i] = -__builtin_inff(); s[i] = 0.0f; }
}

// ---------- init out accumulator to broadcast bias ----------
__global__ void k_init_out(float* __restrict__ out, const float* __restrict__ b,
                           long total, int HF) {
  long i = (long)blockIdx.x * blockDim.x + threadIdx.x;
  if (i < total) out[i] = b[i % HF];
}

// ---------- edge pass 1: segment max of leakyrelu scores ----------
__global__ void k_edge_max(const int* __restrict__ src, const int* __restrict__ dst,
                           const float* __restrict__ el, const float* __restrict__ er,
                           float* __restrict__ m, long E, int H) {
  long i = (long)blockIdx.x * blockDim.x + threadIdx.x;
  if (i >= E * H) return;
  long e = i / H;
  int h = (int)(i - e * H);
  int sn = src[e], dn = dst[e];
  float v = el[(long)sn * H + h] + er[(long)dn * H + h];
  v = v > 0.0f ? v : NEG_SLOPE * v;
  atomicMaxF(&m[(long)dn * H + h], v);
}

// ---------- edge pass 2: exp(e - m[dst]); segment sum ----------
__global__ void k_edge_exp(const int* __restrict__ src, const int* __restrict__ dst,
                           const float* __restrict__ el, const float* __restrict__ er,
                           const float* __restrict__ m, float* __restrict__ ex,
                           float* __restrict__ s, long E, int H) {
  long i = (long)blockIdx.x * blockDim.x + threadIdx.x;
  if (i >= E * H) return;
  long e = i / H;
  int h = (int)(i - e * H);
  int sn = src[e], dn = dst[e];
  float v = el[(long)sn * H + h] + er[(long)dn * H + h];
  v = v > 0.0f ? v : NEG_SLOPE * v;
  float w = __expf(v - m[(long)dn * H + h]);
  ex[i] = w;
  atomicAdd(&s[(long)dn * H + h], w);
}

// ---------- edge pass 3: out[dst,h,:] += alpha * z[src,h,:] (float4 chunks) ----------
__global__ void k_edge_aggregate(const int* __restrict__ src, const int* __restrict__ dst,
                                 const float* __restrict__ ex, const float* __restrict__ s,
                                 const float* __restrict__ z, float* __restrict__ out,
                                 long E, int H, int F) {
  int fc = F >> 2;                 // float4 chunks per head
  long total = E * (long)H * fc;
  long i = (long)blockIdx.x * blockDim.x + threadIdx.x;
  if (i >= total) return;
  long e = i / ((long)H * fc);
  int r = (int)(i - e * (long)H * fc);
  int h = r / fc;
  int c = r - h * fc;
  int sn = src[e], dn = dst[e];
  float alpha = ex[e * H + h] / s[(long)dn * H + h];
  const float4 zv = *(const float4*)(z + ((long)sn * H + h) * F + (c << 2));
  float* op = out + ((long)dn * H + h) * F + (c << 2);
  atomicAdd(op + 0, alpha * zv.x);
  atomicAdd(op + 1, alpha * zv.y);
  atomicAdd(op + 2, alpha * zv.z);
  atomicAdd(op + 3, alpha * zv.w);
}

// ---------- host ----------
struct LayerCfg { int fin, fout, h; };
static const LayerCfg LCFG[5] = {{128,64,4},{256,64,2},{128,64,2},{128,64,1},{64,4,1}};

static inline int cdiv(long a, long b) { return (int)((a + b - 1) / b); }

extern "C" void kernel_launch(void* const* d_in, const int* in_sizes, int n_in,
                              void* d_out, int out_size, void* d_ws, size_t ws_size,
                              hipStream_t stream) {
  const float* x   = (const float*)d_in[0];
  const int*   src = (const int*)d_in[1];
  const int*   dst = (const int*)d_in[2];
  long N = in_sizes[0] / 128;
  long E = in_sizes[1];

  // carve workspace (256B aligned slices)
  char* p = (char*)d_ws;
  auto carve = [&](size_t bytes) -> void* {
    void* r = (void*)p;
    p += (bytes + 255) & ~(size_t)255;
    return r;
  };
  __bf16* A16 = (__bf16*)carve((size_t)N * 256 * sizeof(__bf16));   // activations bf16
  float*  Z   = (float*) carve((size_t)N * 256 * sizeof(float));    // gemm output
  float*  OUT = (float*) carve((size_t)N * 256 * sizeof(float));    // aggregation accum
  float*  EL  = (float*) carve((size_t)N * 4 * sizeof(float));
  float*  ER  = (float*) carve((size_t)N * 4 * sizeof(float));
  float*  M   = (float*) carve((size_t)N * 4 * sizeof(float));
  float*  S   = (float*) carve((size_t)N * 4 * sizeof(float));
  float*  EX  = (float*) carve((size_t)E * 4 * sizeof(float));
  __bf16* WP  = (__bf16*)carve((size_t)256 * 256 * sizeof(__bf16)); // packed W (reused)

  const int TB = 256;
  const float* curIn = x;

  for (int L = 0; L < 5; ++L) {
    int fin = LCFG[L].fin, F = LCFG[L].fout, H = LCFG[L].h;
    int Nc = H * F;
    int NcPad = (Nc + 15) & ~15;
    int NTilesPad = NcPad / 16;
    const float* W  = (const float*)d_in[3 + 4 * L + 0];
    const float* al = (const float*)d_in[3 + 4 * L + 1];
    const float* ar = (const float*)d_in[3 + 4 * L + 2];
    const float* b  = (const float*)d_in[3 + 4 * L + 3];
    float* outBuf = (L == 4) ? (float*)d_out : OUT;

    // 1. activations -> bf16 (fused ELU except layer 1)
    long nA = N * (long)fin;
    k_cvt_bf16<<<cdiv(nA, TB), TB, 0, stream>>>(curIn, A16, nA, L > 0 ? 1 : 0);

    // 2. pack/convert W to column-major bf16 (zero-padded columns)
    long nW = (long)NcPad * fin;
    k_pack_w<<<cdiv(nW, TB), TB, 0, stream>>>(W, WP, fin, Nc, NcPad);

    // 3. WMMA GEMM  Z = A16 x WP   (one wave per block; exact grid)
    int Mtiles = (int)(N / 16);
    if ((NTilesPad & 3) == 0) {
      int nGroups = NTilesPad / 4;
      k_gemm_wmma<4><<<Mtiles * nGroups, 32, 0, stream>>>(A16, WP, Z, fin, Nc, nGroups);
    } else {
      int nGroups = NTilesPad;
      k_gemm_wmma<1><<<Mtiles * nGroups, 32, 0, stream>>>(A16, WP, Z, fin, Nc, nGroups);
    }

    // 4. per-node attention scores
    k_node_scores<<<cdiv(N * H, TB), TB, 0, stream>>>(Z, al, ar, EL, ER, N, H, F);

    // 5. init segment max/sum and output accumulator (bias broadcast)
    k_init_ms<<<cdiv(N * H, TB), TB, 0, stream>>>(M, S, N * H);
    k_init_out<<<cdiv(N * (long)Nc, TB), TB, 0, stream>>>(outBuf, b, N * (long)Nc, Nc);

    // 6-8. edge softmax + weighted aggregation
    k_edge_max<<<cdiv(E * H, TB), TB, 0, stream>>>(src, dst, EL, ER, M, E, H);
    k_edge_exp<<<cdiv(E * H, TB), TB, 0, stream>>>(src, dst, EL, ER, M, EX, S, E, H);
    long aggThreads = E * (long)H * (F / 4);
    k_edge_aggregate<<<cdiv(aggThreads, TB), TB, 0, stream>>>(src, dst, EX, S, Z, outBuf,
                                                              E, H, F);
    curIn = outBuf;
  }
}
